// attention_Layer_2800318677590
// MI455X (gfx1250) — matrix operand order repaired and checked by reading, not compiled, not run on hardware
//
#include <hip/hip_runtime.h>
#include <hip/hip_bf16.h>

typedef __attribute__((ext_vector_type(16))) __bf16 v16bf;
typedef __attribute__((ext_vector_type(8)))  __bf16 v8bf;
typedef __attribute__((ext_vector_type(2)))  __bf16 v2bf;
typedef __attribute__((ext_vector_type(4)))  __bf16 v4bf;
typedef __attribute__((ext_vector_type(8)))  float  v8f;

__device__ __forceinline__ v8f wmma_bf16(v16bf a, v16bf b, v8f c) {
  return __builtin_amdgcn_wmma_f32_16x16x32_bf16(
      false, a, false, b, (short)0, c, false, false);
}

// A-fragment (16x32 bf16, wave32 layout): lane's 16 elems are one contiguous
// 16-elem run: [k0+16*half .. +15].
__device__ __forceinline__ v16bf load_a_frag(const __bf16* base) {
  union { v16bf v; v8bf h[2]; } f;
  f.h[0] = *(const v8bf*)(base);
  f.h[1] = *(const v8bf*)(base + 8);
  return f.v;
}

// -----------------------------------------------------------------------------
// Prep: f32 -> bf16 elementwise (n multiple of 4)
// -----------------------------------------------------------------------------
__global__ void cvt_bf16_kernel(const float* __restrict__ in,
                                __bf16* __restrict__ out, int n) {
  int i = (blockIdx.x * 256 + threadIdx.x) * 4;
  if (i < n) {
    float4 f = *(const float4*)(in + i);
    v4bf o = {(__bf16)f.x, (__bf16)f.y, (__bf16)f.z, (__bf16)f.w};
    *(v4bf*)(out + i) = o;
  }
}

// -----------------------------------------------------------------------------
// Prep: batched transpose + convert, no integer division.
// in[b, r, c] (f32) -> out[b, c, r] (bf16)
// grid: (R/256, C, batch); writes coalesced along r.
// -----------------------------------------------------------------------------
__global__ void transpose_cvt_kernel(const float* __restrict__ in,
                                     __bf16* __restrict__ out, int R, int C) {
  const size_t base = (size_t)blockIdx.z * R * C;
  const int r = blockIdx.x * 256 + threadIdx.x;
  const int c = blockIdx.y;
  out[base + (size_t)c * R + r] = (__bf16)in[base + (size_t)r * C + c];
}

// -----------------------------------------------------------------------------
// Per-head projection, LDS-free, 32x32 per wave (4 WMMA / k-chunk).
// Out[b,h,s,e] (or [b,h,e,s] if transposeOut) = sum_d X[b,s,d]*WT[h,e,d] + bias
// Block: 256 thr (8 waves). Block tile: 128(s) x 64(e).
// Wave w: rows 32*(w&3)..+31 (two 16-slabs), cols 32*(w>>2)..+31 (two subtiles).
// -----------------------------------------------------------------------------
__global__ void qkv_proj_kernel(const __bf16* __restrict__ X,   // [B,S,D] bf16
                                const __bf16* __restrict__ WT,  // [H,E,D] bf16
                                const float*  __restrict__ bias,// [H,E]
                                __bf16* __restrict__ Out,
                                int S_, int D_, int H_, int E_, int transposeOut) {
  const int tid   = threadIdx.x;
  const int wave  = tid >> 5;
  const int lane  = tid & 31;
  const int half  = lane >> 4;
  const int lcol  = lane & 15;
  const int rquad = wave & 3;   // 32-row group
  const int cpair = wave >> 2;  // 32-col group (0 or 1)

  const int b  = blockIdx.z;
  const int h  = blockIdx.y;
  const int s0 = blockIdx.x * 128;

  const __bf16* Arow0 =
      X + ((size_t)b * S_ + s0 + 32 * rquad + lcol) * D_ + 16 * half;
  const __bf16* Arow1 = Arow0 + (size_t)16 * D_;
  const __bf16* Brow0 =
      WT + ((size_t)h * E_ + 32 * cpair + lcol) * D_ + 16 * half;
  const __bf16* Brow1 = Brow0 + (size_t)16 * D_;

  v8f acc00 = {}; v8f acc01 = {}; v8f acc10 = {}; v8f acc11 = {};
  for (int k0 = 0; k0 < D_; k0 += 32) {
    v16bf a0 = load_a_frag(Arow0 + k0);
    v16bf a1 = load_a_frag(Arow1 + k0);
    v16bf b0 = *(const v16bf*)(Brow0 + k0);
    v16bf b1 = *(const v16bf*)(Brow1 + k0);
    acc00 = wmma_bf16(a0, b0, acc00);
    acc01 = wmma_bf16(a0, b1, acc01);
    acc10 = wmma_bf16(a1, b0, acc10);
    acc11 = wmma_bf16(a1, b1, acc11);
  }

  const int c0 = 32 * cpair + lcol, c1 = c0 + 16;
  const float bb0 = bias[h * E_ + c0], bb1 = bias[h * E_ + c1];
  if (!transposeOut) {
    __bf16* outp = Out + (((size_t)b * H_ + h) * S_ + s0) * E_;
#pragma unroll
    for (int r = 0; r < 8; ++r) {
      int row0 = 32 * rquad + r + 8 * half;
      int row1 = row0 + 16;
      outp[(size_t)row0 * E_ + c0] = (__bf16)(acc00[r] + bb0);
      outp[(size_t)row0 * E_ + c1] = (__bf16)(acc01[r] + bb1);
      outp[(size_t)row1 * E_ + c0] = (__bf16)(acc10[r] + bb0);
      outp[(size_t)row1 * E_ + c1] = (__bf16)(acc11[r] + bb1);
    }
  } else {  // [b,h,e,s]
    __bf16* outp = Out + (((size_t)b * H_ + h) * E_) * S_;
#pragma unroll
    for (int r = 0; r < 8; ++r) {
      int row0 = s0 + 32 * rquad + r + 8 * half;
      int row1 = row0 + 16;
      outp[(size_t)c0 * S_ + row0] = (__bf16)(acc00[r] + bb0);
      outp[(size_t)c1 * S_ + row0] = (__bf16)(acc01[r] + bb1);
      outp[(size_t)c0 * S_ + row1] = (__bf16)(acc10[r] + bb0);
      outp[(size_t)c1 * S_ + row1] = (__bf16)(acc11[r] + bb1);
    }
  }
}

// -----------------------------------------------------------------------------
// Flash attention. Q,K fragments straight from global (K-dim = E contiguous);
// V pre-transposed [B,H,E,S] so PV B-fragments are contiguous too.
// Softmax via LDS; P stored in fragment-ready swizzled layout (b128 reads).
// -----------------------------------------------------------------------------
__global__ void flash_attn_kernel(const __bf16* __restrict__ Qh,   // [B,H,S,E]
                                  const __bf16* __restrict__ Kh,   // [B,H,S,E]
                                  const __bf16* __restrict__ VhT,  // [B,H,E,S]
                                  __bf16* __restrict__ Cat,        // [B,S,H*E]
                                  int S_, int H_, int E_) {
  __shared__ float  Sc[64][64];
  __shared__ __bf16 Pfrag[4][32][32];  // [rslab][lane][ch*16 + i]
  __shared__ float  m_s[64], l_s[64], alpha_s[64];
  __shared__ float  pmax[64][4], psum[64][4];

  const int tid   = threadIdx.x;
  const int wave  = tid >> 5;
  const int lane  = tid & 31;
  const int half  = lane >> 4;
  const int lcol  = lane & 15;
  const int rslab = wave & 3;
  const int cbase = (wave >> 2) * 2;

  const int b  = blockIdx.z;
  const int h  = blockIdx.y;
  const int s0 = blockIdx.x * 64;
  const float scale = 0.125f;  // 1/sqrt(64)

  if (tid < 64) { m_s[tid] = -3.0e38f; l_s[tid] = 0.0f; }

  const size_t headOff = ((size_t)b * H_ + h) * S_;
  const __bf16* Qrow = Qh + (headOff + s0 + 16 * rslab + lcol) * E_;
  const __bf16* Khd  = Kh + headOff * E_;
  const __bf16* Vhd  = VhT + headOff * E_;  // [E, S] for this head

  v16bf qA[2];
#pragma unroll
  for (int ch = 0; ch < 2; ++ch)
    qA[ch] = load_a_frag(Qrow + 32 * ch + 16 * half);

  v8f o0 = {}; v8f o1 = {};
  __syncthreads();

  const int srow = tid >> 2, qq = tid & 3;  // softmax mapping: 4 thr/row
  const int p_rs = srow >> 4, p_lc = srow & 15;

  for (int j0 = 0; j0 < S_; j0 += 64) {
    // ---- scores tile: S = Q @ K^T.  B[k=e][n=t] = Kh[t][e] (e contiguous) ----
    const __bf16* Kr0 = Khd + (size_t)(j0 + 16 * cbase + lcol) * E_ + 16 * half;
    const __bf16* Kr1 = Kr0 + (size_t)16 * E_;
    v8f sa = {}; v8f sb = {};
#pragma unroll
    for (int ch = 0; ch < 2; ++ch) {
      v16bf bk0 = *(const v16bf*)(Kr0 + 32 * ch);
      v16bf bk1 = *(const v16bf*)(Kr1 + 32 * ch);
      sa = wmma_bf16(qA[ch], bk0, sa);
      sb = wmma_bf16(qA[ch], bk1, sb);
    }
#pragma unroll
    for (int r = 0; r < 8; ++r) {
      int row = 16 * rslab + r + 8 * half;
      Sc[row][16 * cbase + lcol]       = sa[r] * scale;
      Sc[row][16 * (cbase + 1) + lcol] = sb[r] * scale;
    }
    __syncthreads();

    // ---- online softmax ----
    float mx = -3.0e38f;
#pragma unroll
    for (int c = 0; c < 16; ++c) mx = fmaxf(mx, Sc[srow][16 * qq + c]);
    pmax[srow][qq] = mx;
    __syncthreads();
    if (qq == 0) {
      float mnew = fmaxf(fmaxf(pmax[srow][0], pmax[srow][1]),
                         fmaxf(pmax[srow][2], pmax[srow][3]));
      mnew = fmaxf(mnew, m_s[srow]);
      alpha_s[srow] = __expf(m_s[srow] - mnew);
      m_s[srow] = mnew;
    }
    __syncthreads();
    {
      float mnew = m_s[srow];
      float ssum = 0.0f;
#pragma unroll
      for (int cpair = 0; cpair < 8; ++cpair) {
        int t = 16 * qq + 2 * cpair;                 // even key index
        float p0 = __expf(Sc[srow][t]     - mnew);
        float p1 = __expf(Sc[srow][t + 1] - mnew);
        ssum += p0 + p1;
        // fragment-ready A layout for P: invert (half, i) from k=t
        int ch = t >> 5, tl = t & 31;
        int hlf = tl >> 4;
        int i   = tl & 15;                           // even
        v2bf pp = {(__bf16)p0, (__bf16)p1};
        *(v2bf*)&Pfrag[p_rs][16 * hlf + p_lc][ch * 16 + i] = pp;
      }
      psum[srow][qq] = ssum;
    }
    __syncthreads();
    if (qq == 0)
      l_s[srow] = alpha_s[srow] * l_s[srow] +
                  psum[srow][0] + psum[srow][1] + psum[srow][2] + psum[srow][3];
    __syncthreads();

    // ---- rescale running O, then O += P @ V ----
#pragma unroll
    for (int r = 0; r < 8; ++r) {
      float al = alpha_s[16 * rslab + r + 8 * half];
      o0[r] *= al; o1[r] *= al;
    }
    const __bf16* Vr0 = Vhd + (size_t)(16 * cbase + lcol) * S_ + j0 + 16 * half;
    const __bf16* Vr1 = Vr0 + (size_t)16 * S_;
#pragma unroll
    for (int ch = 0; ch < 2; ++ch) {
      v16bf pa  = *(const v16bf*)&Pfrag[rslab][lane][ch * 16];
      v16bf bv0 = *(const v16bf*)(Vr0 + 32 * ch);
      v16bf bv1 = *(const v16bf*)(Vr1 + 32 * ch);
      o0 = wmma_bf16(pa, bv0, o0);
      o1 = wmma_bf16(pa, bv1, o1);
    }
    __syncthreads();
  }

  const int HE = H_ * E_;
  __bf16* catp = Cat + ((size_t)b * S_ + s0) * HE + h * E_;
#pragma unroll
  for (int r = 0; r < 8; ++r) {
    int row = 16 * rslab + r + 8 * half;
    float inv = 1.0f / l_s[row];
    catp[(size_t)row * HE + 16 * cbase + lcol]       = (__bf16)(o0[r] * inv);
    catp[(size_t)row * HE + 16 * (cbase + 1) + lcol] = (__bf16)(o1[r] * inv);
  }
}

// -----------------------------------------------------------------------------
// Output projection + bias + residual, LDS-free, 32x32 per wave.
// Xres[m,n] = Qin[m,n] + bo[n] + sum_k Cat[m,k]*WoT[n,k]
// Block tile: 128(m) x 64(n).
// -----------------------------------------------------------------------------
__global__ void out_proj_kernel(const __bf16* __restrict__ Cat,  // [M,D] bf16
                                const __bf16* __restrict__ WoT,  // [D,D] bf16 (n,k)
                                const float*  __restrict__ bo,
                                const float*  __restrict__ Qin,  // [M,D] f32
                                float* __restrict__ Xres, int D_) {
  const int tid   = threadIdx.x;
  const int wave  = tid >> 5;
  const int lane  = tid & 31;
  const int half  = lane >> 4;
  const int lcol  = lane & 15;
  const int rquad = wave & 3;
  const int cpair = wave >> 2;

  const int m0 = blockIdx.x * 128;
  const int n0 = blockIdx.y * 64;

  const __bf16* Arow0 = Cat + (size_t)(m0 + 32 * rquad + lcol) * D_ + 16 * half;
  const __bf16* Arow1 = Arow0 + (size_t)16 * D_;
  const __bf16* Brow0 = WoT + (size_t)(n0 + 32 * cpair + lcol) * D_ + 16 * half;
  const __bf16* Brow1 = Brow0 + (size_t)16 * D_;

  v8f acc00 = {}; v8f acc01 = {}; v8f acc10 = {}; v8f acc11 = {};
  for (int k0 = 0; k0 < D_; k0 += 32) {
    v16bf a0 = load_a_frag(Arow0 + k0);
    v16bf a1 = load_a_frag(Arow1 + k0);
    v16bf b0 = *(const v16bf*)(Brow0 + k0);
    v16bf b1 = *(const v16bf*)(Brow1 + k0);
    acc00 = wmma_bf16(a0, b0, acc00);
    acc01 = wmma_bf16(a0, b1, acc01);
    acc10 = wmma_bf16(a1, b0, acc10);
    acc11 = wmma_bf16(a1, b1, acc11);
  }

  const int c0 = n0 + 32 * cpair + lcol, c1 = c0 + 16;
  const float bb0 = bo[c0], bb1 = bo[c1];
#pragma unroll
  for (int r = 0; r < 8; ++r) {
    int row0 = m0 + 32 * rquad + r + 8 * half;
    int row1 = row0 + 16;
    Xres[(size_t)row0 * D_ + c0] = acc00[r] + bb0 + Qin[(size_t)row0 * D_ + c0];
    Xres[(size_t)row0 * D_ + c1] = acc01[r] + bb1 + Qin[(size_t)row0 * D_ + c1];
    Xres[(size_t)row1 * D_ + c0] = acc10[r] + bb0 + Qin[(size_t)row1 * D_ + c0];
    Xres[(size_t)row1 * D_ + c1] = acc11[r] + bb1 + Qin[(size_t)row1 * D_ + c1];
  }
}

// -----------------------------------------------------------------------------
// LayerNorm, block per row, float4 I/O. D_ == 4*blockDim.x
// -----------------------------------------------------------------------------
__global__ void layernorm_kernel(const float* __restrict__ X,
                                 const float* __restrict__ gamma,
                                 const float* __restrict__ beta,
                                 float* __restrict__ Out, int D_) {
  __shared__ float red[256];
  const int row = blockIdx.x;
  const float4* x4 = (const float4*)(X + (size_t)row * D_);
  const float4  f  = x4[threadIdx.x];

  float s = f.x + f.y + f.z + f.w;
  red[threadIdx.x] = s;
  __syncthreads();
  for (int off = 128; off > 0; off >>= 1) {
    if (threadIdx.x < off) red[threadIdx.x] += red[threadIdx.x + off];
    __syncthreads();
  }
  float mu = red[0] * (1.0f / D_);
  __syncthreads();

  float dx = f.x - mu, dy = f.y - mu, dz = f.z - mu, dw = f.w - mu;
  red[threadIdx.x] = dx * dx + dy * dy + dz * dz + dw * dw;
  __syncthreads();
  for (int off = 128; off > 0; off >>= 1) {
    if (threadIdx.x < off) red[threadIdx.x] += red[threadIdx.x + off];
    __syncthreads();
  }
  float rstd = rsqrtf(red[0] * (1.0f / D_) + 1e-5f);

  const float4 g  = ((const float4*)gamma)[threadIdx.x];
  const float4 bt = ((const float4*)beta)[threadIdx.x];
  float4 o;
  o.x = dx * rstd * g.x + bt.x;
  o.y = dy * rstd * g.y + bt.y;
  o.z = dz * rstd * g.z + bt.z;
  o.w = dw * rstd * g.w + bt.w;
  ((float4*)(Out + (size_t)row * D_))[threadIdx.x] = o;
}

// -----------------------------------------------------------------------------
extern "C" void kernel_launch(void* const* d_in, const int* in_sizes, int n_in,
                              void* d_out, int out_size, void* d_ws, size_t ws_size,
                              hipStream_t stream) {
  (void)in_sizes; (void)n_in; (void)out_size; (void)ws_size;

  const float* q     = (const float*)d_in[0];
  const float* k     = (const float*)d_in[1];
  const float* v     = (const float*)d_in[2];
  const float* Wq    = (const float*)d_in[3];
  const float* bq    = (const float*)d_in[4];
  const float* Wk    = (const float*)d_in[5];
  const float* bk    = (const float*)d_in[6];
  const float* Wv    = (const float*)d_in[7];
  const float* bv    = (const float*)d_in[8];
  const float* Wo    = (const float*)d_in[9];
  const float* bo    = (const float*)d_in[10];
  const float* gamma = (const float*)d_in[11];
  const float* beta  = (const float*)d_in[12];
  float* out = (float*)d_out;

  const int B = 2, S = 2048, D = 1024, H = 16, E = 64;
  const size_t NX = (size_t)B * S * D;     // 4 Mi elems
  const size_t NW = (size_t)H * D * E;     // 1 Mi elems
  const size_t NH = (size_t)B * H * S * E; // 4 Mi elems

  // Workspace carve-up (~80 MB):
  __bf16* qbf = (__bf16*)d_ws;             // [B,S,D]
  __bf16* kbf = qbf + NX;
  __bf16* vbf = kbf + NX;
  __bf16* WqT = vbf + NX;                  // [H,E,D]
  __bf16* WkT = WqT + NW;
  __bf16* WvT = WkT + NW;
  __bf16* WoT = WvT + NW;                  // [D,D] (n-major)
  __bf16* qh  = WoT + (size_t)D * D;       // [B,H,S,E]
  __bf16* kh  = qh + NH;
  __bf16* vhT = kh + NH;                   // [B,H,E,S]
  __bf16* cat = vhT + NH;                  // [B*S, D]
  float*  xres = (float*)(cat + NX);       // [B*S, D]

  dim3 blk(256);

  // Prep: conversions + weight transposes
  cvt_bf16_kernel<<<dim3((unsigned)(NX / 4 / 256)), blk, 0, stream>>>(q, qbf, (int)NX);
  cvt_bf16_kernel<<<dim3((unsigned)(NX / 4 / 256)), blk, 0, stream>>>(k, kbf, (int)NX);
  cvt_bf16_kernel<<<dim3((unsigned)(NX / 4 / 256)), blk, 0, stream>>>(v, vbf, (int)NX);
  transpose_cvt_kernel<<<dim3(D / 256, E, H), blk, 0, stream>>>(Wq, WqT, D, E);
  transpose_cvt_kernel<<<dim3(D / 256, E, H), blk, 0, stream>>>(Wk, WkT, D, E);
  transpose_cvt_kernel<<<dim3(D / 256, E, H), blk, 0, stream>>>(Wv, WvT, D, E);
  transpose_cvt_kernel<<<dim3(D / 256, D, 1), blk, 0, stream>>>(Wo, WoT, D, D);

  // Projections (V stored transposed for PV fragment contiguity)
  dim3 gproj(S / 128, H, B);
  qkv_proj_kernel<<<gproj, blk, 0, stream>>>(qbf, WqT, bq, qh,  S, D, H, E, 0);
  qkv_proj_kernel<<<gproj, blk, 0, stream>>>(kbf, WkT, bk, kh,  S, D, H, E, 0);
  qkv_proj_kernel<<<gproj, blk, 0, stream>>>(vbf, WvT, bv, vhT, S, D, H, E, 1);

  flash_attn_kernel<<<dim3(S / 64, H, B), blk, 0, stream>>>(qh, kh, vhT, cat, S, H, E);

  out_proj_kernel<<<dim3((B * S) / 128, D / 64), blk, 0, stream>>>(cat, WoT, bo, q, xres, D);

  layernorm_kernel<<<dim3(B * S), blk, 0, stream>>>(xres, gamma, beta, out, D);
}
